// MoEMLP_13907104104861
// MI455X (gfx1250) — matrix operand order repaired and checked by reading, not compiled, not run on hardware
//
#include <hip/hip_runtime.h>

// ---------------- problem constants ----------------
constexpr int NTOK = 8 * 2048;   // 16384 tokens
constexpr int DM   = 1024;       // model dim
constexpr int FF   = 4096;       // ffn dim
constexpr int NE   = 8;          // experts
constexpr int TOPK = 2;
constexpr int CAP  = 2560;       // int(1.25 * 16384 / 8)

// ---------------- vector types ----------------
typedef __attribute__((ext_vector_type(16))) __bf16 v16bf;
typedef __attribute__((ext_vector_type(8)))  __bf16 v8bf;
typedef __attribute__((ext_vector_type(8)))  float  v8f;
typedef __attribute__((ext_vector_type(4)))  int    v4i;

// ---------------- workspace layout (bytes) ----------------
constexpr size_t SZ_W1T  = (size_t)NE * FF * DM * 2;      // bf16 [E][F][D]
constexpr size_t SZ_W2T  = (size_t)NE * DM * FF * 2;      // bf16 [E][D][F]
constexpr size_t SZ_XBUF = (size_t)NE * CAP * DM * 2;     // bf16 [E][CAP][D]
constexpr size_t SZ_H    = (size_t)NE * CAP * FF * 2;     // bf16 [E][CAP][F]
constexpr size_t OFF_W1T  = 0;
constexpr size_t OFF_W2T  = OFF_W1T + SZ_W1T;
constexpr size_t OFF_XBUF = OFF_W2T + SZ_W2T;
constexpr size_t OFF_H    = OFF_XBUF + SZ_XBUF;
constexpr size_t OFF_STOK = OFF_H + SZ_H;                          // int [E*CAP]
constexpr size_t OFF_SGAT = OFF_STOK + (size_t)NE * CAP * 4;       // float [E*CAP]
constexpr size_t OFF_EFLT = OFF_SGAT + (size_t)NE * CAP * 4;       // int [N*K]
constexpr size_t OFF_GFLT = OFF_EFLT + (size_t)NTOK * TOPK * 4;    // float [N*K]
constexpr size_t OFF_ASLT = OFF_GFLT + (size_t)NTOK * TOPK * 4;    // int [N*K]
constexpr size_t OFF_WSUM = OFF_ASLT + (size_t)NTOK * TOPK * 4;    // float [N]

// ---------------- CDNA5 async global->LDS copy (guarded) ----------------
#if defined(__has_builtin)
#if __has_builtin(__builtin_amdgcn_global_load_async_to_lds_b128) && \
    __has_builtin(__builtin_amdgcn_s_wait_asynccnt)
#define HAVE_ASYNC_LDS 1
#endif
#endif
#ifndef HAVE_ASYNC_LDS
#define HAVE_ASYNC_LDS 0
#endif

__device__ __forceinline__ void copy16_g2l(const __bf16* g, __bf16* l) {
#if HAVE_ASYNC_LDS
  __builtin_amdgcn_global_load_async_to_lds_b128(
      (__attribute__((address_space(1))) v4i*)(size_t)g,
      (__attribute__((address_space(3))) v4i*)(unsigned)(size_t)l,
      0, 0);
#else
  *(v8bf*)l = *(const v8bf*)g;
#endif
}

__device__ __forceinline__ void stage_wait_and_sync() {
#if HAVE_ASYNC_LDS
  __builtin_amdgcn_s_wait_asynccnt(0);
#endif
  __syncthreads();
}

// ---------------- util ----------------
__global__ void zero_f32(float* __restrict__ p, long n) {
  long i = (long)blockIdx.x * blockDim.x + threadIdx.x;
  long s = (long)gridDim.x * blockDim.x;
  for (; i < n; i += s) p[i] = 0.0f;
}

// ---------------- router: one wave32 per token ----------------
__global__ __launch_bounds__(256)
void router_topk(const float* __restrict__ x, const float* __restrict__ wr,
                 const float* __restrict__ br,
                 int* __restrict__ e_flat, float* __restrict__ g_flat) {
  int token = blockIdx.x * 8 + (threadIdx.x >> 5);
  int lane  = threadIdx.x & 31;
  const float* xr = x + (size_t)token * DM;
  float acc[NE];
#pragma unroll
  for (int e = 0; e < NE; ++e) acc[e] = 0.0f;
  for (int i = 0; i < DM / 32; ++i) {
    int d = lane + 32 * i;
    float v = xr[d];
    const float* w = wr + (size_t)d * NE;
#pragma unroll
    for (int e = 0; e < NE; ++e) acc[e] += v * w[e];
  }
#pragma unroll
  for (int e = 0; e < NE; ++e)
#pragma unroll
    for (int off = 16; off > 0; off >>= 1) acc[e] += __shfl_xor(acc[e], off, 32);
  if (lane == 0) {
    float lg[NE], mx = -1e30f;
#pragma unroll
    for (int e = 0; e < NE; ++e) { lg[e] = acc[e] + br[e]; mx = fmaxf(mx, lg[e]); }
    float s = 0.0f;
#pragma unroll
    for (int e = 0; e < NE; ++e) { lg[e] = __expf(lg[e] - mx); s += lg[e]; }
    float inv = 1.0f / s;
    int i0 = 0; float g0 = -1.0f;
#pragma unroll
    for (int e = 0; e < NE; ++e) { float g = lg[e] * inv; if (g > g0) { g0 = g; i0 = e; } }
    int i1 = -1; float g1 = -1.0f;
#pragma unroll
    for (int e = 0; e < NE; ++e) {
      if (e == i0) continue;
      float g = lg[e] * inv; if (g > g1) { g1 = g; i1 = e; }
    }
    e_flat[token * 2 + 0] = i0; g_flat[token * 2 + 0] = g0;
    e_flat[token * 2 + 1] = i1; g_flat[token * 2 + 1] = g1;
  }
}

// ---------------- deterministic position scan (1 workgroup) ----------------
__global__ __launch_bounds__(1024)
void scan_assign(const int* __restrict__ e_flat, const float* __restrict__ g_flat,
                 int* __restrict__ assign_slot, int* __restrict__ slot_tok,
                 float* __restrict__ slot_gate, float* __restrict__ wsum) {
  __shared__ int hist[32][NE];
  __shared__ int base[NE];
  int t = threadIdx.x, lane = t & 31, wave = t >> 5;
  if (t < NE) base[t] = 0;
  __syncthreads();
  for (int it = 0; it < (NTOK * TOPK) / 1024; ++it) {
    int idx = it * 1024 + t;
    int e = e_flat[idx];
    float g = g_flat[idx];
    unsigned lt = (1u << lane) - 1u;
    int myprefix = 0;
#pragma unroll
    for (int q = 0; q < NE; ++q) {
      unsigned m = (unsigned)__ballot(e == q);
      if (q == e) myprefix = __popc(m & lt);
      if (lane == q) hist[wave][q] = __popc(m);
    }
    __syncthreads();
    if (t < NE) {
      int run = base[t];
      for (int w = 0; w < 32; ++w) { int c = hist[w][t]; hist[w][t] = run; run += c; }
      base[t] = run;
    }
    __syncthreads();
    int pos = hist[wave][e] + myprefix;
    int slot = -1;
    if (pos < CAP) {
      slot = e * CAP + pos;
      slot_tok[slot]  = idx >> 1;
      slot_gate[slot] = g;
      __hip_atomic_fetch_add(&wsum[idx >> 1], g, __ATOMIC_RELAXED, __HIP_MEMORY_SCOPE_AGENT);
    }
    assign_slot[idx] = slot;
    __syncthreads();
  }
}

// ---------------- dispatch: copy x row -> bf16 expert buffer ----------------
__global__ __launch_bounds__(256)
void dispatch_copy(const float* __restrict__ x, const int* __restrict__ assign_slot,
                   __bf16* __restrict__ xbuf) {
  int a = blockIdx.x * 8 + (threadIdx.x >> 5);
  int lane = threadIdx.x & 31;
  int slot = assign_slot[a];
  if (slot < 0) return;
  const float* src = x + (size_t)(a >> 1) * DM;
  __bf16* dst = xbuf + (size_t)slot * DM;
#pragma unroll 4
  for (int i = 0; i < DM / 32; ++i) dst[lane + 32 * i] = (__bf16)src[lane + 32 * i];
}

// ---------------- transpose + convert f32 -> bf16 ----------------
// in: [E][R][C] f32, out: [E][C][R] bf16
__global__ __launch_bounds__(256)
void transpose_cvt(const float* __restrict__ in, __bf16* __restrict__ out, int R, int C) {
  __shared__ float tile[32][33];
  int e = blockIdx.z;
  int c0 = blockIdx.x * 32, r0 = blockIdx.y * 32;
  int tx = threadIdx.x & 31, ty = threadIdx.x >> 5;
  const float* ip = in + (size_t)e * R * C;
  __bf16* op = out + (size_t)e * R * C;
#pragma unroll
  for (int i = 0; i < 4; ++i)
    tile[ty + i * 8][tx] = ip[(size_t)(r0 + ty + i * 8) * C + c0 + tx];
  __syncthreads();
#pragma unroll
  for (int i = 0; i < 4; ++i)
    op[(size_t)(c0 + ty + i * 8) * R + r0 + tx] = (__bf16)tile[tx][ty + i * 8];
}

// ---------------- WMMA fragment helpers ----------------
// A frag (16x32 bf16): lane<16 row=lane, K{0..7,16..23}; lane>=16 row=lane-16, K{8..15,24..31}
__device__ __forceinline__ v16bf load_a_frag(const __bf16* rowbase /* row*32 */, int halfsel) {
  v8bf lo = *(const v8bf*)(rowbase + halfsel * 8);
  v8bf hi = *(const v8bf*)(rowbase + halfsel * 8 + 16);
  return __builtin_shufflevector(lo, hi, 0,1,2,3,4,5,6,7,8,9,10,11,12,13,14,15);
}
// B frag (32x16 bf16): lane<16 col=lane, K{0..7,16..23}; lane>=16 col=lane-16, K{8..15,24..31}
__device__ __forceinline__ v16bf load_b_frag(const __bf16* rowbase, int halfsel) {
  v8bf lo = *(const v8bf*)(rowbase + halfsel * 8);
  v8bf hi = *(const v8bf*)(rowbase + halfsel * 8 + 16);
  return __builtin_shufflevector(lo, hi, 0,1,2,3,4,5,6,7,8,9,10,11,12,13,14,15);
}

// ---------------- shared WMMA mainloop ----------------
// A: [M][KDIM] bf16 (K contiguous), Bt: [N][KDIM] bf16 (K contiguous).
// 128x128 block tile, BK=32, 8 waves (2x4), wave tile 64x32, double-buffered LDS
// fed by async global->LDS (ASYNCcnt) when available.
template <int KDIM>
__device__ __forceinline__ void wmma_mainloop(const __bf16* __restrict__ A,
                                              const __bf16* __restrict__ Bt,
                                              int m0, int n0,
                                              __bf16* lsA, __bf16* lsB,  // [2][128*32] each
                                              v8f acc[4][2]) {
  const int t = threadIdx.x, wave = t >> 5, lane = t & 31;
  const int wm = wave & 1, wn = wave >> 1;
  const int halfsel = lane >> 4, lrow = lane & 15;
  // this thread's two 16B chunks of the 128x32 tile
  const int c0 = t * 2, row0 = c0 >> 2, off0 = (c0 & 3) * 8;
  const int c1 = t * 2 + 1, row1 = c1 >> 2, off1 = (c1 & 3) * 8;

  auto issue = [&](int k0, int buf) {
    __bf16* la = lsA + buf * (128 * 32);
    __bf16* lb = lsB + buf * (128 * 32);
    copy16_g2l(A  + (size_t)(m0 + row0) * KDIM + k0 + off0, la + row0 * 32 + off0);
    copy16_g2l(Bt + (size_t)(n0 + row0) * KDIM + k0 + off0, lb + row0 * 32 + off0);
    copy16_g2l(A  + (size_t)(m0 + row1) * KDIM + k0 + off1, la + row1 * 32 + off1);
    copy16_g2l(Bt + (size_t)(n0 + row1) * KDIM + k0 + off1, lb + row1 * 32 + off1);
  };

  issue(0, 0);
  int cur = 0;
  for (int k0 = 0; k0 < KDIM; k0 += 32) {
    stage_wait_and_sync();                 // own async loads done + all waves arrived
    if (k0 + 32 < KDIM) issue(k0 + 32, cur ^ 1);
    const __bf16* la = lsA + cur * (128 * 32);
    const __bf16* lb = lsB + cur * (128 * 32);
    v16bf af[4], bfr[2];
#pragma unroll
    for (int mi = 0; mi < 4; ++mi)
      af[mi] = load_a_frag(la + (wm * 64 + mi * 16 + lrow) * 32, halfsel);
#pragma unroll
    for (int ni = 0; ni < 2; ++ni)
      bfr[ni] = load_b_frag(lb + (wn * 32 + ni * 16 + lrow) * 32, halfsel);
#pragma unroll
    for (int mi = 0; mi < 4; ++mi)
#pragma unroll
      for (int ni = 0; ni < 2; ++ni)
        acc[mi][ni] = __builtin_amdgcn_wmma_f32_16x16x32_bf16(
            false, af[mi], false, bfr[ni], (short)0, acc[mi][ni], false, false);
    cur ^= 1;
  }
}

// ---------------- GEMM1: h = gelu(xbuf @ w1 + b1), bf16 out ----------------
__global__ __launch_bounds__(256)
void gemm1_gelu(const __bf16* __restrict__ xbuf, const __bf16* __restrict__ w1t,
                const float* __restrict__ b1, __bf16* __restrict__ h) {
  const int e  = blockIdx.z;
  const int m0 = blockIdx.x * 128;
  const int n0 = blockIdx.y * 128;
  const __bf16* A  = xbuf + (size_t)e * CAP * DM;
  const __bf16* Bt = w1t  + (size_t)e * FF  * DM;
  __shared__ __align__(16) __bf16 lsA[2][128 * 32];
  __shared__ __align__(16) __bf16 lsB[2][128 * 32];
  const int wave = threadIdx.x >> 5, lane = threadIdx.x & 31;
  const int wm = wave & 1, wn = wave >> 1;
  const int halfsel = lane >> 4, lrow = lane & 15;
  v8f acc[4][2] = {};
  wmma_mainloop<DM>(A, Bt, m0, n0, &lsA[0][0], &lsB[0][0], acc);
#pragma unroll
  for (int mi = 0; mi < 4; ++mi)
#pragma unroll
    for (int ni = 0; ni < 2; ++ni) {
      int col = n0 + wn * 32 + ni * 16 + lrow;
      float bias = b1[e * FF + col];
#pragma unroll
      for (int j = 0; j < 8; ++j) {
        int row = m0 + wm * 64 + mi * 16 + halfsel * 8 + j;
        float v = acc[mi][ni][j] + bias;
        float g = 0.5f * v * (1.0f + erff(v * 0.70710678118f));   // exact GELU
        h[(size_t)e * CAP * FF + (size_t)row * FF + col] = (__bf16)g;
      }
    }
}

// ---------------- GEMM2: out = h @ w2 + b2, gate-weighted scatter into y ----
__global__ __launch_bounds__(256)
void gemm2_combine(const __bf16* __restrict__ h, const __bf16* __restrict__ w2t,
                   const float* __restrict__ b2, const int* __restrict__ slot_tok,
                   const float* __restrict__ slot_gate, float* __restrict__ y) {
  const int e  = blockIdx.z;
  const int m0 = blockIdx.x * 128;
  const int n0 = blockIdx.y * 128;
  const __bf16* A  = h   + (size_t)e * CAP * FF;
  const __bf16* Bt = w2t + (size_t)e * DM  * FF;
  __shared__ __align__(16) __bf16 lsA[2][128 * 32];
  __shared__ __align__(16) __bf16 lsB[2][128 * 32];
  __shared__ int   stok[128];
  __shared__ float sgat[128];
  const int t = threadIdx.x, wave = t >> 5, lane = t & 31;
  if (t < 128) {
    int slot = e * CAP + m0 + t;
    stok[t] = slot_tok[slot];
    sgat[t] = slot_gate[slot];
  }
  const int wm = wave & 1, wn = wave >> 1;
  const int halfsel = lane >> 4, lrow = lane & 15;
  v8f acc[4][2] = {};
  wmma_mainloop<FF>(A, Bt, m0, n0, &lsA[0][0], &lsB[0][0], acc);
#pragma unroll
  for (int mi = 0; mi < 4; ++mi)
#pragma unroll
    for (int ni = 0; ni < 2; ++ni) {
      int col = n0 + wn * 32 + ni * 16 + lrow;
      float bias = b2[e * DM + col];
#pragma unroll
      for (int j = 0; j < 8; ++j) {
        int r = wm * 64 + mi * 16 + halfsel * 8 + j;
        float g = sgat[r];
        if (g > 0.0f) {
          float v = g * (acc[mi][ni][j] + bias);
          __hip_atomic_fetch_add(y + (size_t)stok[r] * DM + col, v,
                                 __ATOMIC_RELAXED, __HIP_MEMORY_SCOPE_AGENT);
        }
      }
    }
}

// ---------------- normalize by gate sum ----------------
__global__ __launch_bounds__(256)
void normalize_y(float* __restrict__ y, const float* __restrict__ wsum) {
  long i = (long)blockIdx.x * 256 + threadIdx.x;        // over NTOK*DM
  int tok = (int)(i >> 10);
  float ws = wsum[tok];
  if (ws > 0.0f) y[i] = y[i] / fmaxf(ws, 1e-6f);
}

// ---------------- host launcher ----------------
extern "C" void kernel_launch(void* const* d_in, const int* in_sizes, int n_in,
                              void* d_out, int out_size, void* d_ws, size_t ws_size,
                              hipStream_t stream) {
  const float* x   = (const float*)d_in[0];
  const float* wr  = (const float*)d_in[1];
  const float* br  = (const float*)d_in[2];
  const float* w1  = (const float*)d_in[3];
  const float* b1  = (const float*)d_in[4];
  const float* w2  = (const float*)d_in[5];
  const float* b2  = (const float*)d_in[6];
  float* y = (float*)d_out;

  char* ws = (char*)d_ws;
  __bf16* w1t   = (__bf16*)(ws + OFF_W1T);
  __bf16* w2t   = (__bf16*)(ws + OFF_W2T);
  __bf16* xbuf  = (__bf16*)(ws + OFF_XBUF);
  __bf16* hbuf  = (__bf16*)(ws + OFF_H);
  int*    stok  = (int*)  (ws + OFF_STOK);
  float*  sgat  = (float*)(ws + OFF_SGAT);
  int*    eflt  = (int*)  (ws + OFF_EFLT);
  float*  gflt  = (float*)(ws + OFF_GFLT);
  int*    aslt  = (int*)  (ws + OFF_ASLT);
  float*  wsum  = (float*)(ws + OFF_WSUM);

  // zero accumulators
  zero_f32<<<4096, 256, 0, stream>>>(y, (long)NTOK * DM);
  zero_f32<<<64, 256, 0, stream>>>(wsum, (long)NTOK);
  zero_f32<<<64, 256, 0, stream>>>(sgat, (long)NE * CAP);

  // routing
  router_topk<<<NTOK / 8, 256, 0, stream>>>(x, wr, br, eflt, gflt);
  scan_assign<<<1, 1024, 0, stream>>>(eflt, gflt, aslt, stok, sgat, wsum);
  dispatch_copy<<<(NTOK * TOPK) / 8, 256, 0, stream>>>(x, aslt, xbuf);

  // weight convert/transpose to bf16, K-contiguous
  transpose_cvt<<<dim3(FF / 32, DM / 32, NE), 256, 0, stream>>>(w1, w1t, DM, FF);
  transpose_cvt<<<dim3(DM / 32, FF / 32, NE), 256, 0, stream>>>(w2, w2t, FF, DM);

  // grouped GEMMs on WMMA
  gemm1_gelu<<<dim3(CAP / 128, FF / 128, NE), 256, 0, stream>>>(xbuf, w1t, b1, hbuf);
  gemm2_combine<<<dim3(CAP / 128, DM / 128, NE), 256, 0, stream>>>(hbuf, w2t, b2, stok, sgat, y);

  // combine normalization
  normalize_y<<<(NTOK * DM) / 256, 256, 0, stream>>>(y, wsum);
}
